// HunyuanVideo15Attention_4913442587190
// MI455X (gfx1250) — compile-verified, matrix operand order checked by reading, not run on hardware
//
#include <hip/hip_runtime.h>

// ---------------------------------------------------------------------------
// HunyuanVideo attention block for MI455X (gfx1250, wave32, WMMA).
// Precision: bf16 inputs to all matrix units, f32 accumulation
// (V_WMMA_F32_16X16X32_BF16). GEMM A-tiles staged to LDS with
// GLOBAL_LOAD_ASYNC_TO_LDS_B128 (ASYNCcnt double-buffering) when available.
// ---------------------------------------------------------------------------

typedef __bf16 bf16;
typedef __attribute__((ext_vector_type(16))) __bf16 v16bf;
typedef __attribute__((ext_vector_type(8)))  __bf16 v8bf;
typedef __attribute__((ext_vector_type(4)))  __bf16 v4bf;
typedef __attribute__((ext_vector_type(8)))  float  v8f;
typedef __attribute__((ext_vector_type(4)))  float  v4f;
typedef __attribute__((ext_vector_type(4)))  int    v4i;

#define S_VID  2048
#define S_TXT  256
#define S_TOT  2304
#define DMODEL 3072
#define NH     24
#define DH     128

#if defined(__gfx1250__) && __has_builtin(__builtin_amdgcn_global_load_async_to_lds_b128)
#define HAVE_ASYNC_LDS 1
#else
#define HAVE_ASYNC_LDS 0
#endif

// ---------------------------------------------------------------------------
// Fragment helpers (CDNA5 16-bit WMMA layouts, wave32).
// A (16x32, MxK) per ISA 7.12.2: lane L holds row M=L&15; lanes 0-15 carry
// K={0..7,16..23}, lanes 16-31 carry K={8..15,24..31}. Two contiguous 16B
// loads per lane. B (32x16, KxN) mirrored: lane holds column N, same K
// pattern, i.e. rows of W when computing X @ W^T. C/D (16x16): lane L holds
// column N=L&15, rows M = 8*(L>=16) + v for VGPR v.
// ---------------------------------------------------------------------------
__device__ __forceinline__ v16bf load_frag(const bf16* __restrict__ p0, int ld) {
  const int lane = threadIdx.x & 31;
  const bf16* p = p0 + (long)(lane & 15) * ld + ((lane >> 4) << 3);
  v8bf lo = *(const v8bf*)(p);        // K = kh .. kh+7
  v8bf hi = *(const v8bf*)(p + 16);   // K = 16+kh .. 16+kh+7
  return __builtin_shufflevector(lo, hi, 0,1,2,3,4,5,6,7,8,9,10,11,12,13,14,15);
}

__device__ __forceinline__ v8f wmma_bf16(v16bf a, v16bf b, v8f c) {
  return __builtin_amdgcn_wmma_f32_16x16x32_bf16(
      /*neg_a=*/false, a, /*neg_b=*/false, b,
      /*c_mod=*/(short)0, c, /*reuse_a=*/false, /*reuse_b=*/false);
}

#if HAVE_ASYNC_LDS
// Builtin signature (from toolchain diagnostic): param0 is v4i in AS1
// ("__device__"), so pass typed pointers: (v4i AS1*, v4i AS3*, imm, imm).
__device__ __forceinline__ void async_copy16(const bf16* g, bf16* l) {
  __builtin_amdgcn_global_load_async_to_lds_b128(
      (__attribute__((address_space(1))) v4i*)g,
      (__attribute__((address_space(3))) v4i*)l, /*offset=*/0, /*cpol=*/0);
}

template <int N>
__device__ __forceinline__ void wait_asynccnt() {
#if __has_builtin(__builtin_amdgcn_s_wait_asynccnt)
  __builtin_amdgcn_s_wait_asynccnt(N);
#else
  asm volatile("s_wait_asynccnt %0" ::"n"(N));
#endif
}
#endif  // HAVE_ASYNC_LDS

// ---------------------------------------------------------------------------
// f32 -> bf16 conversion (vectorized x4).
// ---------------------------------------------------------------------------
__global__ __launch_bounds__(256)
void cvt_f32_bf16_kernel(const float* __restrict__ in, bf16* __restrict__ out, long n4) {
  long i = (long)blockIdx.x * blockDim.x + threadIdx.x;
  if (i < n4) {
    v4f x = *(const v4f*)(in + 4 * i);
    v4bf y;
    y[0] = (bf16)x[0]; y[1] = (bf16)x[1]; y[2] = (bf16)x[2]; y[3] = (bf16)x[3];
    *(v4bf*)(out + 4 * i) = y;
  }
}

// ---------------------------------------------------------------------------
// C[M][N] (f32) = X[M][K](bf16) @ W[N][K]^T(bf16) + bias[N]
// Block: 256 threads = 8 waves arranged 4(M) x 2(N); wave tile 32x64 =
// 2x4 WMMA accumulators; block tile 128x128; K stepped by 32.
// A tile (128x32 bf16 = 8KB) is async-staged to LDS, double-buffered on
// ASYNCcnt: next tile's copies issue before waiting on the current tile.
// B fragments load direct from global (weights are L2-resident).
// All problem dims are multiples of the tiles (no edge handling needed).
// ---------------------------------------------------------------------------
__global__ __launch_bounds__(256)
void gemm_bf16_kernel(const bf16* __restrict__ X, int ldx,
                      const bf16* __restrict__ W, int ldw,
                      const float* __restrict__ bias,
                      float* __restrict__ C, int ldc, int K) {
  const int wave = threadIdx.x >> 5;
  const int lane = threadIdx.x & 31;
  const int wm = wave >> 1;  // 0..3 along M
  const int wn = wave & 1;   // 0..1 along N
  const long blk_m0 = (long)blockIdx.x * 128;
  const long m0 = blk_m0 + wm * 32;
  const long n0 = (long)blockIdx.y * 128 + wn * 64;

  v8f acc[2][4] = {};

#if HAVE_ASYNC_LDS
  __shared__ __align__(16) bf16 ldsA[2][128 * 32];
  const int t = threadIdx.x;
  // 128x32 bf16 tile = 512 chunks of 16B; 256 threads x 2 chunks each.
  auto stage = [&](int k0, int buf) {
#pragma unroll
    for (int u = 0; u < 2; ++u) {
      const int c = t + u * 256;
      const int row = c >> 2;
      const int seg = (c & 3) * 8;
      async_copy16(X + (blk_m0 + row) * ldx + k0 + seg,
                   &ldsA[buf][row * 32 + seg]);
    }
  };

  stage(0, 0);
  int buf = 0;
  for (int k0 = 0; k0 < K; k0 += 32, buf ^= 1) {
    const bool more = (k0 + 32) < K;
    if (more) {
      stage(k0 + 32, buf ^ 1);       // overlap next tile's DMA with compute
      wait_asynccnt<2>();            // current tile done; next may be in flight
      // Hint the next B K-segment toward the caches (per-lane rows).
      __builtin_prefetch(W + (n0 + lane) * ldw + k0 + 32, 0, 0);
      __builtin_prefetch(W + (n0 + 32 + lane) * ldw + k0 + 32, 0, 0);
    } else {
      wait_asynccnt<0>();
    }
    __syncthreads();

    v16bf a0 = load_frag(&ldsA[buf][(wm * 32) * 32], 32);
    v16bf a1 = load_frag(&ldsA[buf][(wm * 32 + 16) * 32], 32);
    v16bf b[4];
#pragma unroll
    for (int j = 0; j < 4; ++j)
      b[j] = load_frag(W + (n0 + 16 * j) * ldw + k0, ldw);
#pragma unroll
    for (int j = 0; j < 4; ++j) {
      acc[0][j] = wmma_bf16(a0, b[j], acc[0][j]);
      acc[1][j] = wmma_bf16(a1, b[j], acc[1][j]);
    }
    __syncthreads();  // all waves done reading before buf is overwritten
  }
#else
  for (int k0 = 0; k0 < K; k0 += 32) {
    v16bf a0 = load_frag(X + m0 * ldx + k0, ldx);
    v16bf a1 = load_frag(X + (m0 + 16) * ldx + k0, ldx);
    v16bf b[4];
#pragma unroll
    for (int j = 0; j < 4; ++j)
      b[j] = load_frag(W + (n0 + 16 * j) * ldw + k0, ldw);
#pragma unroll
    for (int j = 0; j < 4; ++j) {
      acc[0][j] = wmma_bf16(a0, b[j], acc[0][j]);
      acc[1][j] = wmma_bf16(a1, b[j], acc[1][j]);
    }
  }
#endif

  const int nl = lane & 15;
  const int mb = (lane >> 4) << 3;
#pragma unroll
  for (int j = 0; j < 4; ++j) {
    const long col = n0 + 16 * j + nl;
    const float bv = bias ? bias[col] : 0.0f;
#pragma unroll
    for (int i = 0; i < 2; ++i) {
#pragma unroll
      for (int v = 0; v < 8; ++v)
        C[(m0 + 16 * i + mb + v) * ldc + col] = acc[i][j][v] + bv;
    }
  }
}

// ---------------------------------------------------------------------------
// RMSNorm(q,k) + interleaved RoPE (video rows only) + pack to bf16.
// One wave per (sequence position, head); 4 elements/lane, full-wave32
// shuffle reduction for sum-of-squares.
// Outputs: Q,K as [h][s][128]; V transposed as [h][128][s] so attention's
// PV B-fragments load contiguously along the key axis.
// ---------------------------------------------------------------------------
__global__ __launch_bounds__(256)
void norm_rope_pack_kernel(const float* __restrict__ qkv_vid,   // [2048][9216]
                           const float* __restrict__ qkv_enc,   // [256][9216]
                           const float* __restrict__ cos_t,     // [2048][64]
                           const float* __restrict__ sin_t,     // [2048][64]
                           const float* __restrict__ gq,  const float* __restrict__ gk,
                           const float* __restrict__ gaq, const float* __restrict__ gak,
                           bf16* __restrict__ Qo, bf16* __restrict__ Ko,
                           bf16* __restrict__ Vt) {
  const int w = blockIdx.x * 8 + (threadIdx.x >> 5);
  const int lane = threadIdx.x & 31;
  const int s = w / NH;
  const int h = w - s * NH;
  const bool enc = (s < S_TXT);
  const float* row = enc ? (qkv_enc + (long)s * (3 * DMODEL))
                         : (qkv_vid + (long)(s - S_TXT) * (3 * DMODEL));
  const int d0 = lane << 2;

  v4f qv = *(const v4f*)(row + h * DH + d0);
  v4f kv = *(const v4f*)(row + DMODEL + h * DH + d0);
  v4f vv = *(const v4f*)(row + 2 * DMODEL + h * DH + d0);

  float sq = qv[0]*qv[0] + qv[1]*qv[1] + qv[2]*qv[2] + qv[3]*qv[3];
  float sk = kv[0]*kv[0] + kv[1]*kv[1] + kv[2]*kv[2] + kv[3]*kv[3];
#pragma unroll
  for (int m = 1; m < 32; m <<= 1) {
    sq += __shfl_xor(sq, m, 32);
    sk += __shfl_xor(sk, m, 32);
  }
  const float rq = rsqrtf(sq * (1.0f / DH) + 1e-5f);
  const float rk = rsqrtf(sk * (1.0f / DH) + 1e-5f);
  const float* gquse = enc ? gaq : gq;
  const float* gkuse = enc ? gak : gk;

  float q[4], k[4];
#pragma unroll
  for (int t = 0; t < 4; ++t) {
    q[t] = qv[t] * rq * gquse[d0 + t];
    k[t] = kv[t] * rk * gkuse[d0 + t];
  }

  if (!enc) {  // RoPE on video rows: pairs (even, odd), pair index d/2
    const int pos = s - S_TXT;
    const float* cp = cos_t + (long)pos * (DH / 2);
    const float* sp = sin_t + (long)pos * (DH / 2);
    const int p0 = lane << 1;
    const float c0 = cp[p0],     s0 = sp[p0];
    const float c1 = cp[p0 + 1], s1 = sp[p0 + 1];
    float t0 = q[0]*c0 - q[1]*s0, t1 = q[1]*c0 + q[0]*s0;
    float t2 = q[2]*c1 - q[3]*s1, t3 = q[3]*c1 + q[2]*s1;
    q[0]=t0; q[1]=t1; q[2]=t2; q[3]=t3;
    t0 = k[0]*c0 - k[1]*s0; t1 = k[1]*c0 + k[0]*s0;
    t2 = k[2]*c1 - k[3]*s1; t3 = k[3]*c1 + k[2]*s1;
    k[0]=t0; k[1]=t1; k[2]=t2; k[3]=t3;
  }

  const long qoff = ((long)h * S_TOT + s) * DH + d0;
  v4bf qb, kb;
#pragma unroll
  for (int t = 0; t < 4; ++t) { qb[t] = (bf16)q[t]; kb[t] = (bf16)k[t]; }
  *(v4bf*)(Qo + qoff) = qb;
  *(v4bf*)(Ko + qoff) = kb;
#pragma unroll
  for (int t = 0; t < 4; ++t)
    Vt[((long)h * DH + d0 + t) * S_TOT + s] = (bf16)vv[t];
}

// ---------------------------------------------------------------------------
// Flash attention, one head per blockIdx.y, 64 query rows per block
// (16 rows per wave x 4 waves). 32 keys per iteration:
//   QK^T : 2 logit tiles x 4 K-chunks = 8 WMMAs
//   PV   : P(16x32) x V(32x128)       = 8 WMMAs
// P is re-laid out C/D -> A-fragment through a per-wave LDS tile.
// Next K tile is prefetched per-lane (one global_prefetch_b8 covers the
// whole 32x128 tile: one 256B row per lane).
// ---------------------------------------------------------------------------
__global__ __launch_bounds__(128)
void attn_kernel(const bf16* __restrict__ Q, const bf16* __restrict__ Km,
                 const bf16* __restrict__ Vt, bf16* __restrict__ attn) {
  __shared__ __align__(16) bf16 plds[4][16 * 32];
  const int wave = threadIdx.x >> 5;
  const int lane = threadIdx.x & 31;
  const int h = blockIdx.y;
  const long q0 = (long)blockIdx.x * 64 + wave * 16;
  const bf16* Qh = Q  + (long)h * S_TOT * DH;
  const bf16* Kh = Km + (long)h * S_TOT * DH;
  const bf16* Vh = Vt + (long)h * DH * S_TOT;

  v16bf qf[4];
#pragma unroll
  for (int c = 0; c < 4; ++c)
    qf[c] = load_frag(Qh + q0 * DH + 32 * c, DH);

  v8f o[8] = {};
  float mrow[8], lrow[8];
#pragma unroll
  for (int v = 0; v < 8; ++v) { mrow[v] = -1e30f; lrow[v] = 0.0f; }

  const float scale = 0.088388347648318447f;  // 1/sqrt(128)
  const int nl = lane & 15;
  const int mb = (lane >> 4) << 3;

  for (int kb = 0; kb < S_TOT; kb += 32) {
    if (kb + 32 < S_TOT)  // per-lane: one row of the next 32x128 K tile each
      __builtin_prefetch(Kh + (long)(kb + 32 + lane) * DH, 0, 0);

    v8f s0 = {}, s1 = {};
#pragma unroll
    for (int c = 0; c < 4; ++c) {
      v16bf b0 = load_frag(Kh + (long)kb * DH + 32 * c, DH);
      v16bf b1 = load_frag(Kh + (long)(kb + 16) * DH + 32 * c, DH);
      s0 = wmma_bf16(qf[c], b0, s0);
      s1 = wmma_bf16(qf[c], b1, s1);
    }

    // Online softmax: each logit row lives across one 16-lane half.
    float alpha[8];
#pragma unroll
    for (int v = 0; v < 8; ++v) {
      const float l0 = s0[v] * scale, l1 = s1[v] * scale;
      float mx = fmaxf(l0, l1);
#pragma unroll
      for (int msk = 1; msk < 16; msk <<= 1)
        mx = fmaxf(mx, __shfl_xor(mx, msk, 32));
      const float mnew = fmaxf(mrow[v], mx);
      alpha[v] = __expf(mrow[v] - mnew);
      mrow[v] = mnew;
      const float p0 = __expf(l0 - mnew);
      const float p1 = __expf(l1 - mnew);
      s0[v] = p0; s1[v] = p1;
      float rs = p0 + p1;
#pragma unroll
      for (int msk = 1; msk < 16; msk <<= 1)
        rs += __shfl_xor(rs, msk, 32);
      lrow[v] = lrow[v] * alpha[v] + rs;
    }
#pragma unroll
    for (int f = 0; f < 8; ++f)
#pragma unroll
      for (int v = 0; v < 8; ++v)
        o[f][v] *= alpha[v];

    // C/D layout -> A-fragment layout via per-wave LDS tile (16x32 bf16).
    __syncthreads();
#pragma unroll
    for (int v = 0; v < 8; ++v) {
      plds[wave][(mb + v) * 32 + nl]      = (bf16)s0[v];
      plds[wave][(mb + v) * 32 + 16 + nl] = (bf16)s1[v];
    }
    __syncthreads();
    v16bf pf = load_frag(&plds[wave][0], 32);

#pragma unroll
    for (int f = 0; f < 8; ++f) {
      v16bf bv = load_frag(Vh + (long)(f * 16) * S_TOT + kb, S_TOT);
      o[f] = wmma_bf16(pf, bv, o[f]);
    }
  }

#pragma unroll
  for (int v = 0; v < 8; ++v) {
    const float inv = 1.0f / lrow[v];
    const long srow = q0 + mb + v;
#pragma unroll
    for (int f = 0; f < 8; ++f)
      attn[srow * DMODEL + h * DH + f * 16 + nl] = (bf16)(o[f][v] * inv);
  }
}

// ---------------------------------------------------------------------------
// Host orchestration.
// ---------------------------------------------------------------------------
extern "C" void kernel_launch(void* const* d_in, const int* in_sizes, int n_in,
                              void* d_out, int out_size, void* d_ws, size_t ws_size,
                              hipStream_t stream) {
  (void)in_sizes; (void)n_in; (void)out_size; (void)ws_size;
  const float* hidden = (const float*)d_in[0];
  const float* enc_h  = (const float*)d_in[1];
  const float* cos_t  = (const float*)d_in[2];
  const float* sin_t  = (const float*)d_in[3];
  const float* w_qkv  = (const float*)d_in[4];
  const float* w_add  = (const float*)d_in[5];
  const float* b_add  = (const float*)d_in[6];
  const float* gq     = (const float*)d_in[7];
  const float* gk     = (const float*)d_in[8];
  const float* gaq    = (const float*)d_in[9];
  const float* gak    = (const float*)d_in[10];
  const float* w_out  = (const float*)d_in[11];
  const float* b_out  = (const float*)d_in[12];
  const float* w_addo = (const float*)d_in[13];
  const float* b_addo = (const float*)d_in[14];
  float* out = (float*)d_out;

  // Workspace bump allocator (256B aligned).
  char* wsp = (char*)d_ws;
  auto alloc = [&](size_t bytes) -> void* {
    void* p = (void*)wsp;
    wsp += (bytes + 255) & ~(size_t)255;
    return p;
  };
  bf16* Xv      = (bf16*)alloc((size_t)S_VID * DMODEL * 2);
  bf16* Xt      = (bf16*)alloc((size_t)S_TXT * DMODEL * 2);
  bf16* Wqkv    = (bf16*)alloc((size_t)3 * DMODEL * DMODEL * 2);
  bf16* Wadd    = (bf16*)alloc((size_t)3 * DMODEL * DMODEL * 2);
  bf16* Wout    = (bf16*)alloc((size_t)DMODEL * DMODEL * 2);
  bf16* Waddout = (bf16*)alloc((size_t)DMODEL * DMODEL * 2);
  float* qkv_f  = (float*)alloc((size_t)S_VID * 3 * DMODEL * 4);
  float* eqkv_f = (float*)alloc((size_t)S_TXT * 3 * DMODEL * 4);
  bf16* Qb      = (bf16*)alloc((size_t)NH * S_TOT * DH * 2);
  bf16* Kb      = (bf16*)alloc((size_t)NH * S_TOT * DH * 2);
  bf16* Vtb     = (bf16*)alloc((size_t)NH * DH * S_TOT * 2);
  bf16* attnb   = (bf16*)alloc((size_t)S_TOT * DMODEL * 2);

  auto cvt = [&](const float* src, bf16* dst, long n) {
    long n4 = n >> 2;
    int grid = (int)((n4 + 255) / 256);
    cvt_f32_bf16_kernel<<<grid, 256, 0, stream>>>(src, dst, n4);
  };
  cvt(hidden, Xv, (long)S_VID * DMODEL);
  cvt(enc_h, Xt, (long)S_TXT * DMODEL);
  cvt(w_qkv, Wqkv, (long)3 * DMODEL * DMODEL);
  cvt(w_add, Wadd, (long)3 * DMODEL * DMODEL);
  cvt(w_out, Wout, (long)DMODEL * DMODEL);
  cvt(w_addo, Waddout, (long)DMODEL * DMODEL);

  // QKV projections (video: no bias; text: +b_add).
  gemm_bf16_kernel<<<dim3(S_VID / 128, (3 * DMODEL) / 128), 256, 0, stream>>>(
      Xv, DMODEL, Wqkv, DMODEL, nullptr, qkv_f, 3 * DMODEL, DMODEL);
  gemm_bf16_kernel<<<dim3(S_TXT / 128, (3 * DMODEL) / 128), 256, 0, stream>>>(
      Xt, DMODEL, Wadd, DMODEL, b_add, eqkv_f, 3 * DMODEL, DMODEL);

  // RMSNorm + RoPE + pack to head-major bf16 (V transposed).
  norm_rope_pack_kernel<<<(S_TOT * NH) / 8, 256, 0, stream>>>(
      qkv_f, eqkv_f, cos_t, sin_t, gq, gk, gaq, gak, Qb, Kb, Vtb);

  // Flash attention over the concatenated [text(256) | video(2048)] sequence.
  attn_kernel<<<dim3(S_TOT / 64, NH), 128, 0, stream>>>(Qb, Kb, Vtb, attnb);

  // Output projections: video rows -> out[0..], text rows -> out[S_VID*D ..].
  gemm_bf16_kernel<<<dim3(S_VID / 128, DMODEL / 128), 256, 0, stream>>>(
      attnb + (long)S_TXT * DMODEL, DMODEL, Wout, DMODEL, b_out, out, DMODEL, DMODEL);
  gemm_bf16_kernel<<<dim3(S_TXT / 128, DMODEL / 128), 256, 0, stream>>>(
      attnb, DMODEL, Waddout, DMODEL, b_addo, out + (long)S_VID * DMODEL, DMODEL, DMODEL);
}
